// SVDQuantReplicatedLinear_50105088475491
// MI455X (gfx1250) — matrix operand order repaired and checked by reading, not compiled, not run on hardware
//
#include <hip/hip_runtime.h>
#include <hip/hip_bf16.h>

// ---- problem constants (fixed by the reference) ----
#define TT 4096   // tokens
#define KK 4096   // in_features
#define NN 4096   // out_features
#define RR 32     // low rank
#define GG 64     // quant group size

typedef __attribute__((ext_vector_type(16))) _Float16 v16h;
typedef __attribute__((ext_vector_type(8)))  _Float16 v8h;
typedef __attribute__((ext_vector_type(8)))  float    v8f;
typedef __attribute__((ext_vector_type(2)))  float    v2f;

// ---------------------------------------------------------------------------
// Kernel 1: per-token pass over x.
//  - xs = x * smooth
//  - low-rank partials: xpd[t, 0..31] = xs_row @ proj_down   (LDS reduction)
//  - per-group(64) symmetric int4 quant -> dequant -> f16 x_dq
// ---------------------------------------------------------------------------
__global__ __launch_bounds__(256) void quant_lr_kernel(
    const float* __restrict__ x, const float* __restrict__ smooth,
    const float* __restrict__ pd,       // [K, R]
    _Float16* __restrict__ xdq,         // [T, K] f16 out
    float* __restrict__ xpd)            // [T, R] f32 out
{
  const int t   = blockIdx.x;
  const int tid = threadIdx.x;
  const int k0  = tid * 16;

  const float* xr = x + (size_t)t * KK;

  float v[16];
#pragma unroll
  for (int i = 0; i < 16; ++i) v[i] = xr[k0 + i] * smooth[k0 + i];

  // ---- low-rank partial: acc[r] += xs[k] * pd[k][r] ----
  float acc[RR];
#pragma unroll
  for (int r = 0; r < RR; ++r) acc[r] = 0.0f;
#pragma unroll
  for (int i = 0; i < 16; ++i) {
    const float4* p4 = (const float4*)(pd + (size_t)(k0 + i) * RR);
    const float xv = v[i];
#pragma unroll
    for (int rr = 0; rr < 8; ++rr) {
      float4 c = p4[rr];
      acc[4 * rr + 0] += xv * c.x;
      acc[4 * rr + 1] += xv * c.y;
      acc[4 * rr + 2] += xv * c.z;
      acc[4 * rr + 3] += xv * c.w;
    }
  }

  // ---- per-group absmax across the 4 lanes sharing a group of 64 ----
  float m = 0.0f;
#pragma unroll
  for (int i = 0; i < 16; ++i) m = fmaxf(m, fabsf(v[i]));
  m = fmaxf(m, __shfl_xor(m, 1, 32));
  m = fmaxf(m, __shfl_xor(m, 2, 32));
  const float s  = fmaxf(m * (1.0f / 7.0f), 1e-8f);
  const float is = 1.0f / s;

  v16h h;
#pragma unroll
  for (int i = 0; i < 16; ++i) {
    float q = rintf(v[i] * is);
    q = fminf(fmaxf(q, -8.0f), 7.0f);
    h[i] = (_Float16)(q * s);
  }
  *(v16h*)(xdq + (size_t)t * KK + k0) = h;

  // ---- cross-thread reduction of the 32 low-rank partials ----
  __shared__ float red[RR * 257];
#pragma unroll
  for (int r = 0; r < RR; ++r) red[r * 257 + tid] = acc[r];
  __syncthreads();
  if (tid < RR) {
    float ssum = 0.0f;
    for (int j = 0; j < 256; ++j) ssum += red[tid * 257 + j];
    xpd[(size_t)t * RR + tid] = ssum;
  }
}

// ---------------------------------------------------------------------------
// Kernel 2: dequantize int4 weights -> f16 with per-(n,group) scale.
// ---------------------------------------------------------------------------
__global__ __launch_bounds__(256) void wdq_kernel(
    const int* __restrict__ qw,         // [N, K] int4 values in int
    const float* __restrict__ ws,       // [N, K/G]
    _Float16* __restrict__ wdq)         // [N, K] f16 out
{
  const size_t base = ((size_t)blockIdx.x * 256 + threadIdx.x) * 8;
  const int n = (int)(base >> 12);
  const int g = (int)((base & 4095) >> 6);
  const float s = ws[(size_t)n * (KK / GG) + g];
  v8h h;
#pragma unroll
  for (int i = 0; i < 8; ++i) h[i] = (_Float16)((float)qw[base + i] * s);
  *(v8h*)(wdq + base) = h;
}

// ---------------------------------------------------------------------------
// Kernel 3: main GEMM  out = x_dq @ w_dq^T  (+ xpd @ proj_up + bias)
// Block: 256 threads = 8 wave32 (2 wm x 4 wn), block tile 128(M) x 128(N).
// A slab (128 rows x 32 k, 8KB f16) staged in LDS via the CDNA5 async path
// (global_load_async_to_lds_b128, ASYNCcnt), double-buffered so the fill of
// step i+1 overlaps the 8 v_wmma_f32_16x16x32_f16 of step i. B fragments are
// direct b128 loads (only 2-way reuse, L2-resident).
// Epilogue: rank-32 correction via v_wmma_f32_16x16x4_f32, then bias.
// ---------------------------------------------------------------------------
__global__ __launch_bounds__(256) void svdq_gemm_kernel(
    const _Float16* __restrict__ xdq,   // [T, K]
    const _Float16* __restrict__ wdq,   // [N, K]
    const float* __restrict__ xpd,      // [T, R]
    const float* __restrict__ pu,       // [R, N]
    const float* __restrict__ bias,     // [N]
    float* __restrict__ out)            // [T, N]
{
  __shared__ _Float16 shA[2][128][32];  // 2 x 8KB double buffer

  const int tid  = threadIdx.x;
  const int lane = tid & 31;
  const int wid  = tid >> 5;           // 0..7
  const int wm   = wid >> 2;           // 0..1  (M dir)
  const int wn   = wid & 3;            // 0..3  (N dir)
  const int lr16 = lane & 15;
  const int half = lane >> 4;          // 0 or 1

  const int m_blk  = blockIdx.y * 128;
  const int m_base = m_blk + wm * 64;
  const int n_base = blockIdx.x * 128 + wn * 32;

  // per-thread async-fill slice: 2 chunks of 16B (rows r and r+64)
  const int fr = tid >> 2;             // 0..63
  const int fc = (tid & 3) * 8;        // halves within the 32-wide slab

  v8f acc[4][2];
#pragma unroll
  for (int mt = 0; mt < 4; ++mt)
#pragma unroll
    for (int nt = 0; nt < 2; ++nt) acc[mt][nt] = (v8f){0,0,0,0,0,0,0,0};

  const _Float16* brow[2];
#pragma unroll
  for (int nt = 0; nt < 2; ++nt)
    brow[nt] = wdq + (size_t)(n_base + nt * 16 + lr16) * KK;

  // ---- async fill of one 128x32 A slab into LDS buffer `buf` at k ----
  auto issue_fill = [&](int buf, int k) {
    const _Float16* g0 = xdq + (size_t)(m_blk + fr) * KK + k + fc;
    const _Float16* g1 = xdq + (size_t)(m_blk + fr + 64) * KK + k + fc;
    unsigned l0 = (unsigned)(size_t)&shA[buf][fr][fc];
    unsigned l1 = (unsigned)(size_t)&shA[buf][fr + 64][fc];
    asm volatile("global_load_async_to_lds_b128 %0, %1, off"
                 :: "v"(l0), "v"((unsigned long long)(size_t)g0) : "memory");
    asm volatile("global_load_async_to_lds_b128 %0, %1, off"
                 :: "v"(l1), "v"((unsigned long long)(size_t)g1) : "memory");
  };

  const int nsteps = KK / 32;
  issue_fill(0, 0);

  for (int i = 0; i < nsteps; ++i) {
    // my slab writes are done; barrier makes everyone's writes visible
    asm volatile("s_wait_asynccnt 0x0" ::: "memory");
    __syncthreads();
    if (i + 1 < nsteps) issue_fill((i + 1) & 1, (i + 1) * 32);

    const int b = i & 1;
    const int k = i * 32;

    v16h a[4], bb[2];
#pragma unroll
    for (int mt = 0; mt < 4; ++mt) {
      const int row = wm * 64 + mt * 16 + lr16;
      v8h lo = *(const v8h*)&shA[b][row][half * 8];
      v8h hi = *(const v8h*)&shA[b][row][16 + half * 8];
#pragma unroll
      for (int j = 0; j < 8; ++j) { a[mt][j] = lo[j]; a[mt][j + 8] = hi[j]; }
    }
#pragma unroll
    for (int nt = 0; nt < 2; ++nt) {
      const v8h* p = (const v8h*)(brow[nt] + k + half * 8);
      v8h lo = p[0], hi = p[2];
#pragma unroll
      for (int j = 0; j < 8; ++j) { bb[nt][j] = lo[j]; bb[nt][j + 8] = hi[j]; }
    }
#pragma unroll
    for (int mt = 0; mt < 4; ++mt)
#pragma unroll
      for (int nt = 0; nt < 2; ++nt)
        acc[mt][nt] = __builtin_amdgcn_wmma_f32_16x16x32_f16(
            false, a[mt], false, bb[nt], (short)0, acc[mt][nt], false, false);
  }

  // ---- low-rank epilogue: acc += xpd(16x32) @ proj_up(32x16) via f32 WMMA ----
  // B frags hoisted per nt, A frags per mt (no redundant reloads).
  v2f b2s[2][8];
#pragma unroll
  for (int nt = 0; nt < 2; ++nt) {
    const int n = n_base + nt * 16 + lr16;
#pragma unroll
    for (int kk = 0; kk < 8; ++kk) {
      b2s[nt][kk][0] = pu[(size_t)(4 * kk + 2 * half + 0) * NN + n];
      b2s[nt][kk][1] = pu[(size_t)(4 * kk + 2 * half + 1) * NN + n];
    }
  }
#pragma unroll
  for (int mt = 0; mt < 4; ++mt) {
    const int row = m_base + mt * 16 + lr16;
    v2f a2s[8];
#pragma unroll
    for (int kk = 0; kk < 8; ++kk) {
      a2s[kk][0] = xpd[(size_t)row * RR + 4 * kk + 2 * half + 0];
      a2s[kk][1] = xpd[(size_t)row * RR + 4 * kk + 2 * half + 1];
    }
#pragma unroll
    for (int nt = 0; nt < 2; ++nt) {
      v8f c = acc[mt][nt];
#pragma unroll
      for (int kk = 0; kk < 8; ++kk)
        c = __builtin_amdgcn_wmma_f32_16x16x4_f32(
            false, a2s[kk], false, b2s[nt][kk], (short)0, c, false, false);
      acc[mt][nt] = c;
    }
  }

  // ---- bias + store (C/D layout: VGPR i -> row ..+i+8*half, col n) ----
#pragma unroll
  for (int nt = 0; nt < 2; ++nt) {
    const int n  = n_base + nt * 16 + lr16;
    const float bv = bias[n];
#pragma unroll
    for (int mt = 0; mt < 4; ++mt) {
#pragma unroll
      for (int i = 0; i < 8; ++i) {
        const int row = m_base + mt * 16 + i + 8 * half;
        out[(size_t)row * NN + n] = acc[mt][nt][i] + bv;
      }
    }
  }
}

// ---------------------------------------------------------------------------
extern "C" void kernel_launch(void* const* d_in, const int* in_sizes, int n_in,
                              void* d_out, int out_size, void* d_ws, size_t ws_size,
                              hipStream_t stream) {
  const float* x      = (const float*)d_in[0];   // [T,K]
  const int*   qw     = (const int*)d_in[1];     // [N,K]
  const float* wsc    = (const float*)d_in[2];   // [N,K/G]
  const float* pd     = (const float*)d_in[3];   // [K,R]
  const float* pu     = (const float*)d_in[4];   // [R,N]
  const float* smooth = (const float*)d_in[5];   // [K]
  const float* bias   = (const float*)d_in[6];   // [N]
  float* out = (float*)d_out;

  char* wsp = (char*)d_ws;
  _Float16* xdq = (_Float16*)wsp;                                   // 32 MB
  _Float16* wdq = (_Float16*)(wsp + (size_t)TT * KK * sizeof(_Float16));
  float*    xpd = (float*)(wsp + (size_t)(TT + NN) * KK * sizeof(_Float16)); // 512 KB

  quant_lr_kernel<<<TT, 256, 0, stream>>>(x, smooth, pd, xdq, xpd);
  wdq_kernel<<<((size_t)NN * KK) / (256 * 8), 256, 0, stream>>>(qw, wsc, wdq);
  svdq_gemm_kernel<<<dim3(NN / 128, TT / 128), 256, 0, stream>>>(
      xdq, wdq, xpd, pu, bias, out);
}
